// UpDeformConv_87660282511812
// MI455X (gfx1250) — matrix operand order repaired and checked by reading, not compiled, not run on hardware
//
#include <hip/hip_runtime.h>
#include <hip/hip_bf16.h>

typedef __attribute__((ext_vector_type(16))) _Float16 v16h;
typedef __attribute__((ext_vector_type(8)))  _Float16 v8h;
typedef __attribute__((ext_vector_type(8)))  float    v8f;
typedef __attribute__((ext_vector_type(4)))  float    f4;

#define DHW      32768   // 32*32*32 output voxels
#define CIN      64
#define KT       27
#define NOFF     81
#define NOFF_PAD 96
#define COUT     32

// ---------------------------------------------------------------------------
// CDNA5 async global->LDS helpers (ASYNCcnt-tracked; see CDNA5 ISA)
// ---------------------------------------------------------------------------
__device__ __forceinline__ void async_g2l_b128(unsigned lds_off, const void* gaddr) {
  asm volatile("global_load_async_to_lds_b128 %0, %1, off"
               :: "v"(lds_off), "v"(gaddr)
               : "memory");
}
__device__ __forceinline__ void wait_async0() {
#if __has_builtin(__builtin_amdgcn_s_wait_asynccnt)
  __builtin_amdgcn_s_wait_asynccnt(0);
#else
  asm volatile("s_wait_asynccnt 0" ::: "memory");
#endif
}
__device__ __forceinline__ unsigned lds_off_of(const void* p) {
  // AMDGPU flat pointers to LDS keep the wave-relative LDS byte address in the
  // low 32 bits (ISA 10.2).
  return (unsigned)(uintptr_t)p;
}

// ---------------------------------------------------------------------------
// 1) Upsample x (64,16^3) -> voxel-major xup [32768][64] in f32 and f16
// ---------------------------------------------------------------------------
__global__ __launch_bounds__(256) void k_upsample(const float* __restrict__ x,
                                                  float* __restrict__ xup32,
                                                  _Float16* __restrict__ xup16) {
  int v = blockIdx.x * 256 + threadIdx.x;          // voxel in 32^3
  int w = v & 31, h = (v >> 5) & 31, d = v >> 10;
  int sv = (((d >> 1) * 16) + (h >> 1)) * 16 + (w >> 1);   // source voxel 16^3
  #pragma unroll 4
  for (int c = 0; c < CIN; c++) {
    float val = x[c * 4096 + sv];
    xup32[v * CIN + c] = val;
    xup16[v * CIN + c] = (_Float16)val;
  }
}

// ---------------------------------------------------------------------------
// 2) Weight transforms (tap-major so per-tap tiles are contiguous):
//    w_off (81,64,27) -> wofft f16 [27][96][64] (zero pad o to 96)
//    w     (32,64,27) -> wt    f16 [27][32][64]
// ---------------------------------------------------------------------------
__global__ __launch_bounds__(256) void k_wprep(const float* __restrict__ w_off,
                                               const float* __restrict__ w,
                                               _Float16* __restrict__ wofft,
                                               _Float16* __restrict__ wt) {
  int idx = blockIdx.x * 256 + threadIdx.x;
  if (idx < NOFF_PAD * KT) {
    int o = idx / KT, t = idx % KT;
    for (int c = 0; c < CIN; c++) {
      float val = (o < NOFF) ? w_off[(o * CIN + c) * KT + t] : 0.f;
      wofft[(t * NOFF_PAD + o) * CIN + c] = (_Float16)val;
    }
  }
  if (idx < COUT * KT) {
    int o = idx / KT, t = idx % KT;
    for (int c = 0; c < CIN; c++)
      wt[(t * COUT + o) * CIN + c] = (_Float16)w[(o * CIN + c) * KT + t];
  }
}

// ---------------------------------------------------------------------------
// 3) Offset conv as implicit GEMM -> off_t [voxel][96] f32.
//    One wave per 16-voxel M-tile; N=96 via 6 accumulators; K = tap*64+c.
//    Per-tap 12KB B tile double-buffered in LDS via async load-to-LDS.
// ---------------------------------------------------------------------------
#define OFF_TAP_HALVES (NOFF_PAD * CIN)          // 6144 halves = 12288 B
__global__ __launch_bounds__(256) void k_offset_gemm(const _Float16* __restrict__ xup16,
                                                     const _Float16* __restrict__ wofft,
                                                     const float* __restrict__ b_off,
                                                     float* __restrict__ offt) {
  __shared__ _Float16 shb[2 * OFF_TAP_HALVES] __attribute__((aligned(32)));  // 24576 B

  int wave = (blockIdx.x * 256 + threadIdx.x) >> 5;   // 0..2047
  int lane = threadIdx.x & 31;
  int g = lane >> 4;
  int m = lane & 15;
  int base = wave * 16;
  int voxel = base + m;
  int vw = voxel & 31, vh = (voxel >> 5) & 31, vd = voxel >> 10;

  unsigned lbase = lds_off_of(shb);

  v8f acc[6];
  #pragma unroll
  for (int i = 0; i < 6; i++) acc[i] = {};

  // prologue: stage tap 0 into buffer 0
  for (int k = threadIdx.x; k < 768; k += 256)          // 768 * 16B = 12288 B
    async_g2l_b128(lbase + k * 16, (const char*)wofft + k * 16);
  wait_async0();
  __syncthreads();

  int td = -1, th = -1, tw = -1;
  for (int t = 0; t < KT; t++) {
    int buf = t & 1;
    if (t < KT - 1) {   // prefetch next tap's B tile into the other buffer
      const char* src = (const char*)wofft + (size_t)(t + 1) * (OFF_TAP_HALVES * 2);
      unsigned dst = lbase + (buf ^ 1) * (OFF_TAP_HALVES * 2);
      for (int k = threadIdx.x; k < 768; k += 256)
        async_g2l_b128(dst + k * 16, src + k * 16);
    }

    int dn = vd + td, hn = vh + th, wn = vw + tw;
    bool valid = ((unsigned)dn < 32u) & ((unsigned)hn < 32u) & ((unsigned)wn < 32u);
    int nv = (dn * 32 + hn) * 32 + wn;
    const _Float16* btile = shb + buf * OFF_TAP_HALVES;

    #pragma unroll
    for (int cs = 0; cs < 2; cs++) {
      int cb = cs * 32;
      v8h c0 = {}; v8h c1 = {};
      if (valid) {
        const _Float16* p = xup16 + nv * CIN + cb + g * 16;
        c0 = *(const v8h*)(p);
        c1 = *(const v8h*)(p + 8);
      }
      v16h a = __builtin_shufflevector(c0, c1, 0,1,2,3,4,5,6,7,8,9,10,11,12,13,14,15);
      #pragma unroll
      for (int nb = 0; nb < 6; nb++) {
        v16h b = *(const v16h*)(btile + (nb * 16 + m) * CIN + cb + g * 16);
        acc[nb] = __builtin_amdgcn_wmma_f32_16x16x32_f16(
            false, a, false, b, (short)0, acc[nb], false, false);
      }
    }

    // advance tap coordinates (avoid div/mod)
    if (++tw == 2) { tw = -1; if (++th == 2) { th = -1; ++td; } }

    wait_async0();      // next tile staged
    __syncthreads();    // and everyone done reading the buffer we overwrite next
  }

  #pragma unroll
  for (int nb = 0; nb < 6; nb++) {
    int o = nb * 16 + m;
    float bias = (o < NOFF) ? b_off[o] : 0.f;
    #pragma unroll
    for (int r = 0; r < 8; r++)
      offt[(size_t)(base + g * 8 + r) * NOFF_PAD + o] = acc[nb][r] + bias;
  }
}

// ---------------------------------------------------------------------------
// 4) Fused trilinear sampling + main GEMM -> raw[oc][voxel] (conv + bias).
//    One wave per 16-voxel tile, N=32 (2 accumulators), K = tap*64+c.
//    wt staged in LDS in two async phases (13 + 14 taps, 57344 B buffer).
// ---------------------------------------------------------------------------
#define MAIN_TAP_HALVES (COUT * CIN)             // 2048 halves = 4096 B per tap
__global__ __launch_bounds__(256) void k_main(const float* __restrict__ xup32,
                                              const _Float16* __restrict__ wt,
                                              const float* __restrict__ offt,
                                              const float* __restrict__ bias,
                                              float* __restrict__ raw) {
  __shared__ _Float16 shwt[14 * MAIN_TAP_HALVES] __attribute__((aligned(32)));  // 57344 B

  int wave = (blockIdx.x * 256 + threadIdx.x) >> 5;
  int lane = threadIdx.x & 31;
  int g = lane >> 4;
  int m = lane & 15;
  int base = wave * 16;
  int voxel = base + m;
  float vw = (float)(voxel & 31), vh = (float)((voxel >> 5) & 31), vd = (float)(voxel >> 10);

  unsigned lbase = lds_off_of(shwt);
  const float* op = offt + (size_t)voxel * NOFF_PAD;   // this voxel's 81 offsets

  v8f acc0 = {}; v8f acc1 = {};

  for (int phase = 0; phase < 2; phase++) {
    int t0 = phase ? 13 : 0;
    int ntap = phase ? 14 : 13;

    __syncthreads();   // previous phase fully consumed before overwrite
    int chunks = ntap * 256;                      // ntap*4096/16
    const char* src = (const char*)wt + (size_t)t0 * (MAIN_TAP_HALVES * 2);
    for (int k = threadIdx.x; k < chunks; k += 256)
      async_g2l_b128(lbase + k * 16, src + k * 16);
    wait_async0();
    __syncthreads();

    for (int tt = 0; tt < ntap; tt++) {
      int t = t0 + tt;
      int td = t / 9 - 1, th = (t / 3) % 3 - 1, tw = t % 3 - 1;
      float od = op[t * 3 + 0];
      float oh = op[t * 3 + 1];
      float ow = op[t * 3 + 2];

      float pd = vd + (float)td + od;
      float ph = vh + (float)th + oh;
      float pw = vw + (float)tw + ow;
      float d0f = floorf(pd), h0f = floorf(ph), w0f = floorf(pw);
      float fd = pd - d0f, fh = ph - h0f, fw = pw - w0f;
      int d0 = (int)d0f, h0 = (int)h0f, w0 = (int)w0f;

      int cvi[8]; float cw[8];
      #pragma unroll
      for (int dd = 0; dd < 2; dd++)
        #pragma unroll
        for (int hh = 0; hh < 2; hh++)
          #pragma unroll
          for (int ww = 0; ww < 2; ww++) {
            int di = d0 + dd, hi = h0 + hh, wi = w0 + ww;
            bool ok = ((unsigned)di < 32u) & ((unsigned)hi < 32u) & ((unsigned)wi < 32u);
            int dc = di < 0 ? 0 : (di > 31 ? 31 : di);
            int hc = hi < 0 ? 0 : (hi > 31 ? 31 : hi);
            int wc = wi < 0 ? 0 : (wi > 31 ? 31 : wi);
            float wgt = (dd ? fd : 1.f - fd) * (hh ? fh : 1.f - fh) * (ww ? fw : 1.f - fw);
            int k = (dd * 2 + hh) * 2 + ww;
            cvi[k] = ((dc * 32 + hc) * 32 + wc) * CIN;
            cw[k] = ok ? wgt : 0.f;
          }

      const _Float16* btile = shwt + tt * MAIN_TAP_HALVES;
      #pragma unroll
      for (int cs = 0; cs < 2; cs++) {
        int cb = cs * 32;
        int c0 = cb + g * 16;       // A K-chunk 0 (contiguous 8 channels)
        int c1 = cb + g * 16 + 8;   // A K-chunk 1
        f4 s0a = {}, s0b = {}, s1a = {}, s1b = {};
        #pragma unroll
        for (int k = 0; k < 8; k++) {
          const float* p = xup32 + cvi[k];
          float wgt = cw[k];
          f4 a0 = *(const f4*)(p + c0);
          f4 a1 = *(const f4*)(p + c0 + 4);
          f4 b0 = *(const f4*)(p + c1);
          f4 b1 = *(const f4*)(p + c1 + 4);
          s0a += wgt * a0; s0b += wgt * a1;
          s1a += wgt * b0; s1b += wgt * b1;
        }
        union { v16h v; _Float16 h[16]; } ua;
        #pragma unroll
        for (int i = 0; i < 4; i++) {
          ua.h[i]      = (_Float16)s0a[i];
          ua.h[4 + i]  = (_Float16)s0b[i];
          ua.h[8 + i]  = (_Float16)s1a[i];
          ua.h[12 + i] = (_Float16)s1b[i];
        }
        v16h bA = *(const v16h*)(btile + (m)      * CIN + cb + g * 16);
        v16h bB = *(const v16h*)(btile + (16 + m) * CIN + cb + g * 16);
        acc0 = __builtin_amdgcn_wmma_f32_16x16x32_f16(
            false, ua.v, false, bA, (short)0, acc0, false, false);
        acc1 = __builtin_amdgcn_wmma_f32_16x16x32_f16(
            false, ua.v, false, bB, (short)0, acc1, false, false);
      }
    }
  }

  float bz0 = bias[m], bz1 = bias[16 + m];
  #pragma unroll
  for (int r = 0; r < 8; r++) {
    raw[m * DHW + base + g * 8 + r]        = acc0[r] + bz0;
    raw[(16 + m) * DHW + base + g * 8 + r] = acc1[r] + bz1;
  }
}

// ---------------------------------------------------------------------------
// 5) Per-channel BN statistics
// ---------------------------------------------------------------------------
__global__ __launch_bounds__(256) void k_stats(const float* __restrict__ raw,
                                               float* __restrict__ stats) {
  int oc = blockIdx.x;
  int tid = threadIdx.x;
  float s = 0.f, s2 = 0.f;
  for (int i = tid; i < DHW; i += 256) {
    float v = raw[oc * DHW + i];
    s += v; s2 += v * v;
  }
  __shared__ float sh[256];
  __shared__ float sh2[256];
  sh[tid] = s; sh2[tid] = s2;
  __syncthreads();
  for (int st = 128; st > 0; st >>= 1) {
    if (tid < st) { sh[tid] += sh[tid + st]; sh2[tid] += sh2[tid + st]; }
    __syncthreads();
  }
  if (tid == 0) {
    float mean = sh[0] / (float)DHW;
    float var = sh2[0] / (float)DHW - mean * mean;
    stats[oc * 2]     = mean;
    stats[oc * 2 + 1] = rsqrtf(var + 1e-5f);
  }
}

// ---------------------------------------------------------------------------
// 6) Normalize + affine + ReLU -> d_out  (layout [oc][d][h][w] matches raw)
// ---------------------------------------------------------------------------
__global__ __launch_bounds__(256) void k_final(const float* __restrict__ raw,
                                               const float* __restrict__ stats,
                                               const float* __restrict__ gamma,
                                               const float* __restrict__ beta,
                                               float* __restrict__ out) {
  int idx = blockIdx.x * 256 + threadIdx.x;   // 32*32768
  int oc = idx >> 15;
  float v = (raw[idx] - stats[oc * 2]) * stats[oc * 2 + 1] * gamma[oc] + beta[oc];
  out[idx] = v > 0.f ? v : 0.f;
}

// ---------------------------------------------------------------------------
extern "C" void kernel_launch(void* const* d_in, const int* in_sizes, int n_in,
                              void* d_out, int out_size, void* d_ws, size_t ws_size,
                              hipStream_t stream) {
  const float* x     = (const float*)d_in[0];
  const float* w_off = (const float*)d_in[1];
  const float* b_off = (const float*)d_in[2];
  const float* w     = (const float*)d_in[3];
  const float* b     = (const float*)d_in[4];
  const float* gamma = (const float*)d_in[5];
  const float* beta  = (const float*)d_in[6];
  float* out = (float*)d_out;

  char* ws = (char*)d_ws;
  float*    xup32 = (float*)   (ws + 0);          //  8,388,608 B
  _Float16* xup16 = (_Float16*)(ws + 8388608);    //  4,194,304 B
  _Float16* wofft = (_Float16*)(ws + 12582912);   //    331,776 B  [27][96][64]
  _Float16* wt    = (_Float16*)(ws + 12914688);   //    110,592 B  [27][32][64]
  float*    offt  = (float*)   (ws + 13025280);   // 12,582,912 B  [32768][96]
  float*    raw   = (float*)   (ws + 25608192);   //  4,194,304 B
  float*    stats = (float*)   (ws + 29802496);   //        256 B

  k_upsample   <<<128,  256, 0, stream>>>(x, xup32, xup16);
  k_wprep      <<<11,   256, 0, stream>>>(w_off, w, wofft, wt);
  k_offset_gemm<<<256,  256, 0, stream>>>(xup16, wofft, b_off, offt);
  k_main       <<<256,  256, 0, stream>>>(xup32, wt, offt, b, raw);
  k_stats      <<<32,   256, 0, stream>>>(raw, stats);
  k_final      <<<4096, 256, 0, stream>>>(raw, stats, gamma, beta, out);
}
